// EntropyRegularizedVQ_41549513621535
// MI455X (gfx1250) — matrix operand order repaired and checked by reading, not compiled, not run on hardware
//
#include <hip/hip_runtime.h>

#define NUM_CODES   1024
#define EMBED_DIM   64
#define N_TOKENS    131072
#define BLOCK       256          // 8 waves (wave32)
#define WAVES       8
#define TILE_TOKENS 256          // 32 tokens per wave (2 WMMA tiles)
#define NUM_WG      256
#define CODE_BLOCKS (NUM_CODES / 16)
#define SCORE_BIAS  0.5f         // makes all scores positive: uint order == float order

typedef __attribute__((ext_vector_type(16))) __bf16 v16bf;
typedef __attribute__((ext_vector_type(8)))  __bf16 bf16x8;
typedef __attribute__((ext_vector_type(8)))  float  v8f;

struct __align__(8) bf16x4s { __bf16 a, b, c, d; };

static __device__ __forceinline__ v16bf cat8(bf16x8 lo, bf16x8 hi) {
  return __builtin_shufflevector(lo, hi, 0, 1, 2, 3, 4, 5, 6, 7,
                                         8, 9, 10, 11, 12, 13, 14, 15);
}

static __device__ __forceinline__ unsigned umin32(unsigned a, unsigned b) {
  return a < b ? a : b;   // v_min_u32 (no NaN canonicalization, unlike fminf)
}

// ---------------------------------------------------------------------------
// Kernel 0: zero histogram + sum-of-squares accumulator in workspace
// ---------------------------------------------------------------------------
__global__ void vq_init_kernel(unsigned* __restrict__ counts,
                               float* __restrict__ ssq) {
  int t = blockIdx.x * blockDim.x + threadIdx.x;
  if (t < NUM_CODES) counts[t] = 0u;
  if (t == 0) *ssq = 0.0f;
}

// ---------------------------------------------------------------------------
// Kernel 1: nearest-code assignment via bf16 WMMA + gather/MSE/histogram
//
// Scoring: WMMA computes  score = (esq_k + 0.5) + (-2z).e_k  (bias folded into
// the C-init). All scores are strictly positive, so IEEE float ordering equals
// unsigned ordering of the bit patterns. The low 10 mantissa bits carry no
// information (bf16 inputs), so we overwrite them with the code index:
//   key = (score_bits & ~1023) | code
// and the entire argmin (in-lane and cross-lane butterfly) is v_min_u32.
// ---------------------------------------------------------------------------
__global__ __launch_bounds__(BLOCK) void vq_assign_kernel(
    const float* __restrict__ z_e, const float* __restrict__ cb,
    float* __restrict__ zq_out, float* __restrict__ idx_out,
    unsigned* __restrict__ counts, float* __restrict__ ssq_out) {
  // 128KB codebook(bf16) + 4KB biased |e|^2 + 32KB z-tile(bf16) + 1KB indices
  __shared__ __bf16 s_cb[NUM_CODES * EMBED_DIM];
  __shared__ float  s_esq[NUM_CODES];
  __shared__ __bf16 s_z[TILE_TOKENS * EMBED_DIM];
  __shared__ int    s_idx[TILE_TOKENS];

  const int tid  = threadIdx.x;
  const int lane = tid & 31;
  const int w    = tid >> 5;
  const int m    = lane & 15;   // M (A-row) / N (B-col) lane index
  const int h    = lane >> 4;   // lane half selects K range

  // ---- Phase A: codebook -> LDS as bf16, plus biased |e_k|^2 (f32) ----
  for (int code = tid; code < NUM_CODES; code += BLOCK) {
    const float4* row = (const float4*)(cb + code * EMBED_DIM);
    float ss = 0.0f;
#pragma unroll
    for (int j = 0; j < EMBED_DIM / 4; ++j) {
      float4 v = row[j];
      ss += v.x * v.x + v.y * v.y + v.z * v.z + v.w * v.w;
      bf16x4s p = {(__bf16)v.x, (__bf16)v.y, (__bf16)v.z, (__bf16)v.w};
      ((bf16x4s*)(&s_cb[code * EMBED_DIM]))[j] = p;
    }
    s_esq[code] = ss + SCORE_BIAS;
  }
  __syncthreads();

  float ssq_acc = 0.0f;

  // ---- Phase B: grid-stride over 256-token tiles ----
  for (int base = blockIdx.x * TILE_TOKENS; base < N_TOKENS;
       base += gridDim.x * TILE_TOKENS) {
    // Stage z tile into LDS as bf16, pre-scaled by -2 so the WMMA accumulator
    // directly yields  score = bias + |e|^2 - 2 z.e  (argmin-equiv. to d^2).
    for (int e4 = tid; e4 < TILE_TOKENS * EMBED_DIM / 4; e4 += BLOCK) {
      float4 v = ((const float4*)(z_e + (size_t)base * EMBED_DIM))[e4];
      bf16x4s p = {(__bf16)(-2.0f * v.x), (__bf16)(-2.0f * v.y),
                   (__bf16)(-2.0f * v.z), (__bf16)(-2.0f * v.w)};
      ((bf16x4s*)s_z)[e4] = p;
    }
    __syncthreads();

    // Prefetch next tile (global_prefetch_b8 on gfx1250)
    {
      long nb = (long)base + (long)gridDim.x * TILE_TOKENS;
      if (nb < N_TOKENS)
        __builtin_prefetch(z_e + nb * EMBED_DIM + (size_t)tid * 64, 0, 0);
    }

    // Two A-tiles per wave (tokens w*32+0..15 and w*32+16..31), per the
    // documented 16-bit A layout: lanes 0-15 hold K={0..7,16..23},
    // lanes 16-31 hold K={8..15,24..31}.
    const __bf16* zr0 = s_z + (w * 32 + m) * EMBED_DIM;
    const __bf16* zr1 = zr0 + 16 * EMBED_DIM;
    v16bf a00 = cat8(*(const bf16x8*)(zr0 + h * 8),
                     *(const bf16x8*)(zr0 + 16 + h * 8));
    v16bf a01 = cat8(*(const bf16x8*)(zr0 + 32 + h * 8),
                     *(const bf16x8*)(zr0 + 48 + h * 8));
    v16bf a10 = cat8(*(const bf16x8*)(zr1 + h * 8),
                     *(const bf16x8*)(zr1 + 16 + h * 8));
    v16bf a11 = cat8(*(const bf16x8*)(zr1 + 32 + h * 8),
                     *(const bf16x8*)(zr1 + 48 + h * 8));

    unsigned best0[8], best1[8];
#pragma unroll
    for (int r = 0; r < 8; ++r) { best0[r] = 0xFFFFFFFFu; best1[r] = 0xFFFFFFFFu; }

    // Sweep all 1024 codes, 16 at a time; each B fetch feeds 4 WMMAs.
    for (int blk = 0; blk < CODE_BLOCKS; ++blk) {
      // B (32x16 bf16): lane = code column; lanes 0-15 K={0..7,16..23},
      // lanes 16-31 K={8..15,24..31} -> 16B LDS loads matching A.
      const __bf16* cr = s_cb + (blk * 16 + m) * EMBED_DIM;
      v16bf b0 = cat8(*(const bf16x8*)(cr + h * 8),
                      *(const bf16x8*)(cr + 16 + h * 8));
      v16bf b1 = cat8(*(const bf16x8*)(cr + 32 + h * 8),
                      *(const bf16x8*)(cr + 48 + h * 8));

      const unsigned code = (unsigned)(blk * 16 + m);
      const float    esqb = s_esq[blk * 16 + m];
      v8f ci = {esqb, esqb, esqb, esqb, esqb, esqb, esqb, esqb};

      v8f c0 = __builtin_amdgcn_wmma_f32_16x16x32_bf16(
                   false, a00, false, b0, (short)0, ci, false, false);
      c0     = __builtin_amdgcn_wmma_f32_16x16x32_bf16(
                   false, a01, false, b1, (short)0, c0, false, false);
      v8f c1 = __builtin_amdgcn_wmma_f32_16x16x32_bf16(
                   false, a10, false, b0, (short)0, ci, false, false);
      c1     = __builtin_amdgcn_wmma_f32_16x16x32_bf16(
                   false, a11, false, b1, (short)0, c1, false, false);

#pragma unroll
      for (int r = 0; r < 8; ++r) {
        unsigned k0 = (__float_as_uint(c0[r]) & ~1023u) | code;  // v_and_or_b32
        unsigned k1 = (__float_as_uint(c1[r]) & ~1023u) | code;
        best0[r] = umin32(best0[r], k0);                         // v_min_u32
        best1[r] = umin32(best1[r], k1);
      }
    }

    // Branchless integer-min butterfly across the 16 lanes holding the same
    // token (C layout: slot r, lane l -> token M = r + (l>>4)*8, code N = l&15).
#pragma unroll
    for (int off = 1; off < 16; off <<= 1) {
#pragma unroll
      for (int r = 0; r < 8; ++r) {
        best0[r] = umin32(best0[r], (unsigned)__shfl_xor((int)best0[r], off, 32));
        best1[r] = umin32(best1[r], (unsigned)__shfl_xor((int)best1[r], off, 32));
      }
    }
    if (m == 0) {
#pragma unroll
      for (int r = 0; r < 8; ++r) {
        int t0 = h * 8 + r;
        int i0 = (int)(best0[r] & 1023u);
        int i1 = (int)(best1[r] & 1023u);
        s_idx[w * 32 + t0] = i0;
        s_idx[w * 32 + 16 + t0] = i1;
        idx_out[base + w * 32 + t0] = (float)i0;
        idx_out[base + w * 32 + 16 + t0] = (float)i1;
      }
    }

    // Gather z_q = codebook[idx] in exact f32; 2 lanes per token, 2 passes.
#pragma unroll
    for (int tt = 0; tt < 2; ++tt) {
      int t     = tt * 16 + (lane >> 1);
      int sub   = lane & 1;
      int token = base + w * 32 + t;
      int idx   = s_idx[w * 32 + t];
      const float4* q4 = (const float4*)(cb + (size_t)idx * EMBED_DIM + sub * 32);
      const float4* z4 = (const float4*)(z_e + (size_t)token * EMBED_DIM + sub * 32);
      float4*       o4 = (float4*)(zq_out + (size_t)token * EMBED_DIM + sub * 32);
#pragma unroll
      for (int j = 0; j < 8; ++j) {
        float4 q = q4[j];
        float4 z = z4[j];
        o4[j] = q;  // straight-through forward value == z_q
        float dx = q.x - z.x, dy = q.y - z.y, dz = q.z - z.z, dw = q.w - z.w;
        ssq_acc += dx * dx + dy * dy + dz * dz + dw * dw;
      }
      if (sub == 0) atomicAdd(&counts[idx], 1u);
    }
    __syncthreads();  // protect s_z / s_idx before next tile
  }

  // Wave-reduce the squared-error partial and accumulate once per wave.
#pragma unroll
  for (int off = 16; off >= 1; off >>= 1)
    ssq_acc += __shfl_xor(ssq_acc, off, 32);
  if (lane == 0) atomicAdd(ssq_out, ssq_acc);
}

// ---------------------------------------------------------------------------
// Kernel 2: entropy / perplexity / losses from histogram + ssq
// ---------------------------------------------------------------------------
__global__ __launch_bounds__(1024) void vq_finalize_kernel(
    const unsigned* __restrict__ counts, const float* __restrict__ ssq,
    float* __restrict__ scal_out) {
  __shared__ float red[32];
  int t = threadIdx.x;  // one thread per code
  float p = (float)counts[t] / (float)N_TOKENS + 1e-10f;
  float v = -p * logf(p);
#pragma unroll
  for (int off = 16; off >= 1; off >>= 1) v += __shfl_xor(v, off, 32);
  if ((t & 31) == 0) red[t >> 5] = v;
  __syncthreads();
  if (t < 32) {
    float s = red[t];
#pragma unroll
    for (int off = 16; off >= 1; off >>= 1) s += __shfl_xor(s, off, 32);
    if (t == 0) {
      float entropy = s;
      float cl = *ssq / (float)((size_t)N_TOKENS * EMBED_DIM);
      scal_out[0] = cl;                                   // codebook_loss
      scal_out[1] = 0.25f * cl;                           // commitment_loss
      scal_out[2] = -0.1f * (entropy / logf(1024.0f));    // entropy_loss
      scal_out[3] = expf(entropy);                        // perplexity
    }
  }
}

// ---------------------------------------------------------------------------
extern "C" void kernel_launch(void* const* d_in, const int* in_sizes, int n_in,
                              void* d_out, int out_size, void* d_ws,
                              size_t ws_size, hipStream_t stream) {
  (void)in_sizes; (void)n_in; (void)out_size; (void)ws_size;
  const float* z_e = (const float*)d_in[0];
  const float* cb  = (const float*)d_in[1];

  float* out  = (float*)d_out;
  float* zq   = out;                                   // [N_TOKENS, 64]
  float* idxf = out + (size_t)N_TOKENS * EMBED_DIM;    // [N_TOKENS] (as float)
  float* scal = idxf + N_TOKENS;                       // 4 scalars

  unsigned* counts = (unsigned*)d_ws;                  // [1024]
  float*    ssq    = (float*)((char*)d_ws + NUM_CODES * sizeof(unsigned));

  vq_init_kernel<<<4, 256, 0, stream>>>(counts, ssq);
  vq_assign_kernel<<<NUM_WG, BLOCK, 0, stream>>>(z_e, cb, zq, idxf, counts, ssq);
  vq_finalize_kernel<<<1, 1024, 0, stream>>>(counts, ssq, scal);
}
